// ResidualIntegrationNetworkRK4_14035953124039
// MI455X (gfx1250) — matrix operand order repaired and checked by reading, not compiled, not run on hardware
//
#include <hip/hip_runtime.h>

// CDNA5 / gfx1250 RK4 neural-ODE integrator.
// SEQ=256, BATCH=1024, FEAT=2, HIDDEN=128.
// One workgroup (8 wave32 waves) owns 16 batch rows for the whole sequence.
// Hidden layer [16,128]@[128,128] done with v_wmma_f32_16x16x32_f16:
//   - each wave owns a 16-wide N-slice, W2 B-fragments register-resident (f16)
//   - h1 staged through LDS in WMMA A-fragment-friendly padded layout;
//     all 8 A-fragment ds_load_b128s issued before the WMMA chain so LDS
//     latency overlaps WMMA issue (dscnt counts down instead of 0-waits).
// tanh is branch-free (v_exp/v_rcp); cross-lane reduction is DPP row_shr adds
// (pure VALU), since everything sits on one long sequential dependency chain
// (256 steps x 4 RK4 stages).

typedef _Float16 half8  __attribute__((ext_vector_type(8)));
typedef _Float16 half16 __attribute__((ext_vector_type(16)));
typedef float    float8 __attribute__((ext_vector_type(8)));

union H16 { half16 v; _Float16 e[16]; half8 h[2]; };
union F8  { float8 v; float e[8]; };

#define SEQ   256
#define BATCH 1024
#define HID   128
#define HPAD  136   // padded row stride (halfs): 272B -> conflict-free b128 LDS loads

// Branch-free tanh: tanh(x) = sign(x) * (1 - e) / (1 + e), e = exp(-2|x|).
__device__ __forceinline__ float fast_tanh(float x) {
    float ax = __builtin_fabsf(x);
    float e  = __expf(-2.0f * ax);                       // native v_exp_f32 path
    float r  = (1.0f - e) * __builtin_amdgcn_rcpf(1.0f + e);
    return __builtin_copysignf(r, x);
}

// v_add_f32 with DPP row_shr:N, bound_ctrl=1 (invalid lanes contribute 0).
// After N=1,2,4,8: lane 15 holds sum of lanes 0-15, lane 31 sum of 16-31.
#define DPP_ROW_SHR_ADD(v, N)                                              \
    (v) += __int_as_float(__builtin_amdgcn_update_dpp(                     \
               0, __float_as_int(v), 0x110 | (N), 0xF, 0xF, true))

__global__ __launch_bounds__(256)
void rk4_mlp_kernel(const float* __restrict__ xg,
                    const float* __restrict__ W1g,
                    const float* __restrict__ b1g,
                    const float* __restrict__ W2g,
                    const float* __restrict__ b2g,
                    const float* __restrict__ W3g,
                    const float* __restrict__ b3g,
                    float* __restrict__ outg)
{
    __shared__ __align__(16) _Float16 hbuf[16 * HPAD]; // h1 activations (f16)
    __shared__ float W1s[3 * HID];
    __shared__ float b1s[HID];
    __shared__ float vbuf0[16], vbuf1[16];             // x inputs for this step
    __shared__ float ybuf[16], yinbuf[16];             // integrator state / stage input
    __shared__ float part[8][16];                      // per-wave layer-3 partials
    __shared__ float karr[4][16];                      // k1..k4

    const int tid  = threadIdx.x;
    const int lane = tid & 31;
    const int wave = tid >> 5;              // 0..7 : owns hidden cols [wave*16, wave*16+16)
    const int row0 = blockIdx.x * 16;       // batch rows of this workgroup

    // ---------------- startup: weights on-chip ----------------
    for (int i = tid; i < 3 * HID; i += 256) W1s[i] = W1g[i];
    for (int i = tid; i < HID;     i += 256) b1s[i] = b1g[i];
    if (tid < 16) { ybuf[tid] = 0.0f; yinbuf[tid] = 0.0f; }

    const int   nbase = wave * 16;
    const int   ncol  = nbase + (lane & 15);
    const float b2v   = b2g[ncol];          // per-lane bias for this N column
    const float w3v   = W3g[ncol];          // per-lane W3 weight for this N column
    const float b3v   = b3g[0];

    // Build register-resident f16 B-fragments of W2 (32x16 per K-chunk).
    // B layout (16x16x32 f16): lane n = ncol; element j <-> K = kbase + khalf
    // + (j & 7) + (j >> 3) * 16, khalf = 8*(lane>=16) (matches A-fragment order).
    H16 w2frag[4];
    {
        const int khalf = (lane >> 4) * 8;
        for (int kc = 0; kc < 4; ++kc)
            for (int j = 0; j < 16; ++j)
                w2frag[kc].e[j] = (_Float16)W2g[(kc * 32 + khalf + (j & 7) + (j >> 3) * 16) * HID + ncol];
    }
    __syncthreads();

    // ---------------- sequential RK4 over time ----------------
    for (int t = 0; t < SEQ; ++t) {
        if (tid < 16) {
            const float2* x2 = (const float2*)xg;           // [SEQ][BATCH][2]
            float2 vv = x2[(size_t)t * BATCH + row0 + tid];
            vbuf0[tid] = vv.x; vbuf1[tid] = vv.y;
        }
        __syncthreads();

        for (int stage = 0; stage < 4; ++stage) {
            // ---- phase A: layer 1 (K=3, VALU) -> h1 into LDS as f16 pairs ----
            #pragma unroll
            for (int i = 0; i < 4; ++i) {
                int idx = (tid + i * 256) * 2;  // 16*128 elements, 2 per thread-slot
                int m = idx >> 7, k = idx & 127;
                float v0 = vbuf0[m], v1 = vbuf1[m], yv = yinbuf[m];
                float a0 = v0 * W1s[k]     + v1 * W1s[HID + k]
                         + yv * W1s[2 * HID + k]     + b1s[k];
                float a1 = v0 * W1s[k + 1] + v1 * W1s[HID + k + 1]
                         + yv * W1s[2 * HID + k + 1] + b1s[k + 1];
                union { _Float16 h[2]; unsigned u; } p;
                p.h[0] = (_Float16)fast_tanh(a0);
                p.h[1] = (_Float16)fast_tanh(a1);
                *(unsigned*)(hbuf + m * HPAD + k) = p.u;   // ds_store_b32
            }
            __syncthreads();

            // ---- phase B: layer 2 via WMMA (f16 in, f32 acc) ----
            // A-fragment (16x32 f16): lane m=lane&15 (rows), sel=lane>>4:
            //   VGPR0-3 = K[kbase + sel*8 .. +7], VGPR4-7 = K[kbase+16+sel*8 ..]
            const int m   = lane & 15;
            const int sel = lane >> 4;
            const _Float16* rowp = hbuf + m * HPAD + sel * 8;

            // Issue all 8 ds_load_b128 first so LDS latency overlaps the
            // serialized WMMA accumulator chain.
            H16 afrag[4];
            #pragma unroll
            for (int kc = 0; kc < 4; ++kc) {
                afrag[kc].h[0] = *(const half8*)(rowp + kc * 32);
                afrag[kc].h[1] = *(const half8*)(rowp + kc * 32 + 16);
            }

            F8 acc; acc.v = float8{};
            #pragma unroll
            for (int kc = 0; kc < 4; ++kc)
                acc.v = __builtin_amdgcn_wmma_f32_16x16x32_f16(
                            false, afrag[kc].v, false, w2frag[kc].v,
                            (short)0, acc.v, false, false);

            // ---- layer 3: tanh + [128,1] projection, DPP row reduction ----
            float sums[8];
            #pragma unroll
            for (int r = 0; r < 8; ++r) {
                // C layout: VGPR r, lane l -> row M = r + 8*(l>=16), col N = l%16
                float val = fast_tanh(acc.e[r] + b2v) * w3v;
                DPP_ROW_SHR_ADD(val, 1);
                DPP_ROW_SHR_ADD(val, 2);
                DPP_ROW_SHR_ADD(val, 4);
                DPP_ROW_SHR_ADD(val, 8);
                sums[r] = val;          // lane 15: rows 0..7; lane 31: rows 8..15
            }
            if ((lane & 15) == 15) {
                #pragma unroll
                for (int r = 0; r < 8; ++r)
                    part[wave][sel * 8 + r] = sums[r];     // conflict-free stores
            }
            __syncthreads();

            // ---- phase C: stage bookkeeping (sum 8 wave partials) ----
            if (tid < 16) {
                float kv = b3v;
                #pragma unroll
                for (int w = 0; w < 8; ++w) kv += part[w][tid];
                karr[stage][tid] = kv;
                if (stage == 0)      yinbuf[tid] = ybuf[tid] + 0.5f * kv;
                else if (stage == 1) yinbuf[tid] = ybuf[tid] + 0.5f * kv;
                else if (stage == 2) yinbuf[tid] = ybuf[tid] + kv;
            }
            __syncthreads();
        }

        // ---- RK4 combine + emit ----
        if (tid < 16) {
            float yn = ybuf[tid] + (karr[0][tid] + 2.0f * karr[1][tid]
                     + 2.0f * karr[2][tid] + karr[3][tid]) * (1.0f / 6.0f);
            ybuf[tid]   = yn;
            yinbuf[tid] = yn;
            outg[(size_t)t * BATCH + row0 + tid] = yn;
        }
        // top-of-loop __syncthreads() orders these writes before next phase A
    }
}

extern "C" void kernel_launch(void* const* d_in, const int* in_sizes, int n_in,
                              void* d_out, int out_size, void* d_ws, size_t ws_size,
                              hipStream_t stream) {
    const float* x  = (const float*)d_in[0];
    const float* W1 = (const float*)d_in[1];
    const float* b1 = (const float*)d_in[2];
    const float* W2 = (const float*)d_in[3];
    const float* b2 = (const float*)d_in[4];
    const float* W3 = (const float*)d_in[5];
    const float* b3 = (const float*)d_in[6];
    float* out = (float*)d_out;

    dim3 grid(BATCH / 16);   // 64 workgroups, 16 batch rows each
    dim3 block(256);         // 8 wave32 waves: one per 16-wide hidden N-slice
    hipLaunchKernelGGL(rk4_mlp_kernel, grid, block, 0, stream,
                       x, W1, b1, W2, b2, W3, b3, out);
}
